// ManualLSTMCell_27659589386821
// MI455X (gfx1250) — matrix operand order repaired and checked by reading, not compiled, not run on hardware
//
#include <hip/hip_runtime.h>
#include <hip/hip_bf16.h>

#define B_DIM 4096
#define IN_DIM 1024
#define H_DIM 1024
#define BH ((size_t)B_DIM * H_DIM)

typedef __attribute__((ext_vector_type(16))) __bf16 v16bf;
typedef __attribute__((ext_vector_type(8)))  float  v8f;

// ---------------- fp32 -> bf16 (round-to-nearest-even) conversion ----------------

__device__ __forceinline__ unsigned short f2bf(float f) {
    union { float f; unsigned int u; } v;
    v.f = f;
    unsigned int u = v.u;
    unsigned int r = u + 0x7FFFu + ((u >> 16) & 1u);
    return (unsigned short)(r >> 16);
}

__global__ void cvt_f32_bf16_v4(const float4* __restrict__ src,
                                ushort4* __restrict__ dst, int n4) {
    int i = blockIdx.x * blockDim.x + threadIdx.x;
    int stride = gridDim.x * blockDim.x;
    for (; i < n4; i += stride) {
        float4 f = src[i];
        ushort4 o;
        o.x = f2bf(f.x); o.y = f2bf(f.y); o.z = f2bf(f.z); o.w = f2bf(f.w);
        dst[i] = o;
    }
}

// ---------------- activations ----------------

__device__ __forceinline__ float fast_sigmoid(float z) {
    return 1.0f / (1.0f + __expf(-z));
}

__device__ __forceinline__ float fast_tanh(float x) {
    float a = fabsf(x);
    float e = __expf(-2.0f * a);
    float t = (1.0f - e) / (1.0f + e);
    return copysignf(t, x);
}

// ---------------- fused LSTM cell ----------------
// One wave computes a 32(batch) x 16(hidden) tile for ALL FOUR gates:
//   8 accumulators (4 gates x 2 row sub-tiles), A fragments reused across
//   gates, B fragments reused across row sub-tiles, epilogue fully in
//   registers (no LDS, no barriers).

union AFrag {
    uint4  q[2];   // two 16-byte global loads
    v16bf  v;
};

__global__ __launch_bounds__(256) void lstm_cell_wmma(
    const unsigned short* __restrict__ x_bf,   // B x IN   (bf16)
    const unsigned short* __restrict__ h_bf,   // B x H    (bf16)
    const unsigned short* __restrict__ wx_bf,  // 4 x H x IN, gate-major (i,f,g,o)
    const unsigned short* __restrict__ wh_bf,  // 4 x H x H,  gate-major (i,f,g,o)
    const float* __restrict__ b_i, const float* __restrict__ b_f,
    const float* __restrict__ b_g, const float* __restrict__ b_o,
    const float* __restrict__ c_prev,          // B x H (fp32)
    float* __restrict__ out)                   // [h | c | i | f | g | o], each B*H
{
    const int lane = threadIdx.x & 31;
    const int wv   = threadIdx.x >> 5;                  // wave in block: 0..7
    const int colBase = blockIdx.x << 4;                // 16 hidden cols
    const int rowBase = (blockIdx.y * 8 + wv) << 5;     // 32 batch rows per wave
    const int half = lane >> 4;                         // lane group 0/1
    const int l16  = lane & 15;

    // Per-lane fragment addressing (ISA 7.12.2):
    //  A (16x32 bf16): lanes 0-15 row M=l16, K = {k0..k0+7, k0+16..k0+23};
    //                  lanes 16-31 same rows, K shifted by +8.
    //  B (32x16 bf16): lane = column N=l16, same K order as A
    //                  (K = {k0..k0+7, k0+16..k0+23}, +8 for lanes 16-31) == rows of (out,in) W.
    const int aOff = 8 * half;
    const int bOff = 8 * half;

    v8f acc[4][2];
#pragma unroll
    for (int g = 0; g < 4; ++g)
#pragma unroll
        for (int rt = 0; rt < 2; ++rt)
            acc[g][rt] = (v8f){0.f, 0.f, 0.f, 0.f, 0.f, 0.f, 0.f, 0.f};

    // ---- phase 1: gates += x @ Wx[g]^T  (K = IN = 1024) ----
    {
        const unsigned short* aRow0 = x_bf + (size_t)(rowBase + l16) * IN_DIM;
        const unsigned short* aRow1 = aRow0 + (size_t)16 * IN_DIM;
        const unsigned short* bRow[4];
#pragma unroll
        for (int g = 0; g < 4; ++g)
            bRow[g] = wx_bf + (size_t)g * ((size_t)H_DIM * IN_DIM)
                            + (size_t)(colBase + l16) * IN_DIM;

#pragma unroll 2
        for (int k0 = 0; k0 < IN_DIM; k0 += 32) {
            AFrag a0, a1, b[4];
            a0.q[0] = *reinterpret_cast<const uint4*>(aRow0 + k0 + aOff);
            a0.q[1] = *reinterpret_cast<const uint4*>(aRow0 + k0 + aOff + 16);
            a1.q[0] = *reinterpret_cast<const uint4*>(aRow1 + k0 + aOff);
            a1.q[1] = *reinterpret_cast<const uint4*>(aRow1 + k0 + aOff + 16);
#pragma unroll
            for (int g = 0; g < 4; ++g) {
                b[g].q[0] = *reinterpret_cast<const uint4*>(bRow[g] + k0 + bOff);
                b[g].q[1] = *reinterpret_cast<const uint4*>(bRow[g] + k0 + bOff + 16);
            }
#pragma unroll
            for (int g = 0; g < 4; ++g) {
                acc[g][0] = __builtin_amdgcn_wmma_f32_16x16x32_bf16(
                                false, a0.v, false, b[g].v, (short)0, acc[g][0], false, false);
                acc[g][1] = __builtin_amdgcn_wmma_f32_16x16x32_bf16(
                                false, a1.v, false, b[g].v, (short)0, acc[g][1], false, false);
            }
        }
    }

    // ---- phase 2: gates += h_prev @ Wh[g]^T  (K = H = 1024) ----
    {
        const unsigned short* aRow0 = h_bf + (size_t)(rowBase + l16) * H_DIM;
        const unsigned short* aRow1 = aRow0 + (size_t)16 * H_DIM;
        const unsigned short* bRow[4];
#pragma unroll
        for (int g = 0; g < 4; ++g)
            bRow[g] = wh_bf + (size_t)g * ((size_t)H_DIM * H_DIM)
                            + (size_t)(colBase + l16) * H_DIM;

#pragma unroll 2
        for (int k0 = 0; k0 < H_DIM; k0 += 32) {
            AFrag a0, a1, b[4];
            a0.q[0] = *reinterpret_cast<const uint4*>(aRow0 + k0 + aOff);
            a0.q[1] = *reinterpret_cast<const uint4*>(aRow0 + k0 + aOff + 16);
            a1.q[0] = *reinterpret_cast<const uint4*>(aRow1 + k0 + aOff);
            a1.q[1] = *reinterpret_cast<const uint4*>(aRow1 + k0 + aOff + 16);
#pragma unroll
            for (int g = 0; g < 4; ++g) {
                b[g].q[0] = *reinterpret_cast<const uint4*>(bRow[g] + k0 + bOff);
                b[g].q[1] = *reinterpret_cast<const uint4*>(bRow[g] + k0 + bOff + 16);
            }
#pragma unroll
            for (int g = 0; g < 4; ++g) {
                acc[g][0] = __builtin_amdgcn_wmma_f32_16x16x32_bf16(
                                false, a0.v, false, b[g].v, (short)0, acc[g][0], false, false);
                acc[g][1] = __builtin_amdgcn_wmma_f32_16x16x32_bf16(
                                false, a1.v, false, b[g].v, (short)0, acc[g][1], false, false);
            }
        }
    }

    // ---- fully in-register epilogue ----
    // C/D layout: VGPR j of sub-tile rt -> row rowBase + rt*16 + j + 8*half,
    //             col colBase + l16.
    const int col = colBase + l16;
    const float bi = b_i[col];
    const float bf = b_f[col];
    const float bg = b_g[col];
    const float bo = b_o[col];

#pragma unroll
    for (int rt = 0; rt < 2; ++rt) {
#pragma unroll
        for (int j = 0; j < 8; ++j) {
            const int row = rowBase + rt * 16 + j + 8 * half;
            const size_t idx = (size_t)row * H_DIM + col;

            float iv = fast_sigmoid(acc[0][rt][j] + bi);
            float fv = fast_sigmoid(acc[1][rt][j] + bf);
            float gv = fast_tanh   (acc[2][rt][j] + bg);
            float ov = fast_sigmoid(acc[3][rt][j] + bo);

            float c = fv * c_prev[idx] + iv * gv;
            float h = ov * fast_tanh(c);

            out[idx]           = h;   // h_t
            out[BH + idx]      = c;   // c_t
            out[2 * BH + idx]  = iv;  // i_t
            out[3 * BH + idx]  = fv;  // f_t
            out[4 * BH + idx]  = gv;  // g_t
            out[5 * BH + idx]  = ov;  // o_t
        }
    }
}

// ---------------- host launch ----------------

extern "C" void kernel_launch(void* const* d_in, const int* in_sizes, int n_in,
                              void* d_out, int out_size, void* d_ws, size_t ws_size,
                              hipStream_t stream) {
    const float* x      = (const float*)d_in[0];
    const float* hprev  = (const float*)d_in[1];
    const float* cprev  = (const float*)d_in[2];
    const float* W_ii   = (const float*)d_in[3];
    const float* b_ii   = (const float*)d_in[4];
    const float* W_if   = (const float*)d_in[5];
    const float* b_if   = (const float*)d_in[6];
    const float* W_ig   = (const float*)d_in[7];
    const float* b_ig   = (const float*)d_in[8];
    const float* W_io   = (const float*)d_in[9];
    const float* b_io   = (const float*)d_in[10];
    const float* W_hi   = (const float*)d_in[11];
    const float* W_hf   = (const float*)d_in[12];
    const float* W_hg   = (const float*)d_in[13];
    const float* W_ho   = (const float*)d_in[14];

    // workspace layout (bf16): x | h | Wx[4] | Wh[4]  => 32 MB total
    unsigned short* ws    = (unsigned short*)d_ws;
    unsigned short* x_bf  = ws;
    unsigned short* h_bf  = x_bf + (size_t)B_DIM * IN_DIM;
    unsigned short* wx_bf = h_bf + (size_t)B_DIM * H_DIM;
    unsigned short* wh_bf = wx_bf + 4 * (size_t)H_DIM * IN_DIM;

    const int thr = 256;
    auto cvt = [&](const float* s, unsigned short* d, size_t n) {
        int n4 = (int)(n / 4);
        int blocks = (n4 + thr - 1) / thr;
        if (blocks > 4096) blocks = 4096;
        cvt_f32_bf16_v4<<<blocks, thr, 0, stream>>>(
            (const float4*)s, (ushort4*)d, n4);
    };

    const size_t WXN = (size_t)H_DIM * IN_DIM;
    const size_t WHN = (size_t)H_DIM * H_DIM;

    cvt(x,     x_bf, (size_t)B_DIM * IN_DIM);
    cvt(hprev, h_bf, (size_t)B_DIM * H_DIM);
    cvt(W_ii,  wx_bf + 0 * WXN, WXN);
    cvt(W_if,  wx_bf + 1 * WXN, WXN);
    cvt(W_ig,  wx_bf + 2 * WXN, WXN);
    cvt(W_io,  wx_bf + 3 * WXN, WXN);
    cvt(W_hi,  wh_bf + 0 * WHN, WHN);
    cvt(W_hf,  wh_bf + 1 * WHN, WHN);
    cvt(W_hg,  wh_bf + 2 * WHN, WHN);
    cvt(W_ho,  wh_bf + 3 * WHN, WHN);

    // 64 col tiles x (16 blocks * 8 waves) row units of 32 rows = 4096 rows
    dim3 grid(H_DIM / 16, B_DIM / (8 * 32), 1);   // (64, 16)
    dim3 block(256, 1, 1);                        // 8 waves per block
    lstm_cell_wmma<<<grid, block, 0, stream>>>(
        x_bf, h_bf, wx_bf, wh_bf,
        b_ii, b_if, b_ig, b_io,
        cprev, (float*)d_out);
}
